// AVWGCN_2594160246843
// MI455X (gfx1250) — matrix operand order repaired and checked by reading, not compiled, not run on hardware
//
#include <hip/hip_runtime.h>
#include <stdint.h>

#define NN 4096
#define PA 40   // LDS A pitch in bf16 units: 32 data + 8 pad (20-dword stride)

typedef __attribute__((ext_vector_type(16))) __bf16 v16bf;
typedef __attribute__((ext_vector_type(8)))  float  v8f;
typedef __attribute__((ext_vector_type(4)))  unsigned int u32x4;
typedef __attribute__((ext_vector_type(8)))  int i32x8;
typedef __attribute__((ext_vector_type(4)))  int i32x4;

union BF16Frag { v16bf v; unsigned int u[8]; };

#if __has_builtin(__builtin_amdgcn_tensor_load_to_lds) && __has_builtin(__builtin_amdgcn_s_wait_tensorcnt)
#define USE_TDM 1
#else
#define USE_TDM 0
#endif

__device__ __forceinline__ unsigned short f2bf(float f) {
    unsigned int u = __float_as_uint(f);
    u += 0x7FFFu + ((u >> 16) & 1u);           // round-to-nearest-even
    return (unsigned short)(u >> 16);
}

__device__ __forceinline__ v8f wmma_bf16(v16bf a, v16bf b, v8f c) {
    return __builtin_amdgcn_wmma_f32_16x16x32_bf16(
        false, a, false, b, (short)0, c, false, false);
}

// ---- Tensor Data Mover: load 128-row x 32-col bf16 tile, LDS pitch 40 bf16 --
#if USE_TDM
__device__ __forceinline__ void tdm_load_tile_a(unsigned int ldsOff, const unsigned short* gptr) {
    unsigned long long ga = (unsigned long long)(size_t)gptr;
    u32x4 g0;
    g0[0] = 1u;                                            // count=1 (valid user D#)
    g0[1] = ldsOff;                                        // lds_addr (bytes)
    g0[2] = (unsigned int)ga;                              // global_addr[31:0]
    g0[3] = (unsigned int)((ga >> 32) & 0x1FFFFFFull) | (2u << 30);  // addr[56:32] | type=2
    i32x8 g1;
    g1[0] = (int)((1u << 16)    // data_size = 2 bytes
                | (1u << 20)    // pad_enable
                | (3u << 22)    // pad_interval: after 16 dwords (one 32-elem row)
                | (3u << 25));  // pad_amount: 4 dwords -> 80B LDS pitch == PA
    g1[1] = (int)(4096u << 16); // tensor_dim0 = 4096 (low 16 in bits 63:48)
    g1[2] = (int)(4096u << 16); // tensor_dim0 hi = 0 | tensor_dim1 low 16
    g1[3] = (int)(32u << 16);   // tensor_dim1 hi = 0 | tile_dim0 = 32 elements
    g1[4] = 128;                // tile_dim1 = 128 rows, tile_dim2 = 0
    g1[5] = 4096;               // tensor_dim0_stride = 4096 elements
    g1[6] = 0;
    g1[7] = 0;
    i32x4 z4 = {0, 0, 0, 0};
#if defined(__clang_major__) && (__clang_major__ >= 23)
    i32x8 z8 = {0, 0, 0, 0, 0, 0, 0, 0};
    __builtin_amdgcn_tensor_load_to_lds(g0, g1, z4, z4, z8, 0);
#else
    __builtin_amdgcn_tensor_load_to_lds(g0, g1, z4, z4, 0);
#endif
}
#endif

__device__ __forceinline__ unsigned int lds_off(const void* p) {
    return (unsigned int)(size_t)p;
}

// Fallback cooperative A staging (row-major global -> padded LDS)
__device__ __forceinline__ void copy_tile_a(unsigned short* As, const unsigned short* Ag,
                                            int rowBlock, int kk, int tid) {
#pragma unroll
    for (int i = 0; i < 2; ++i) {
        int idx = tid + i * 256;                 // 512 uint4 total
        int r = idx >> 2, q = idx & 3;
        uint4 d = *(const uint4*)(Ag + (size_t)(rowBlock + r) * NN + kk + q * 8);
        *(uint4*)(As + r * PA + q * 8) = d;
    }
}

// A fragment 16x32 from padded LDS (two ds_load_b128 per fragment)
__device__ __forceinline__ v16bf frag_a_lds(const unsigned short* As, int rowBase, int lane) {
    BF16Frag f;
    int hf = lane >> 4;
    const unsigned short* base = As + (rowBase + (lane & 15)) * PA;
#pragma unroll
    for (int v = 0; v < 8; ++v) {
        int kb = ((v & 4) ? 16 : 0) + hf * 8 + 2 * (v & 3);
        f.u[v] = *(const unsigned int*)(base + kb);
    }
    return f.v;
}

// B fragment 32x16 straight from global pair-packed matrix (dword = rows k,k+1)
__device__ __forceinline__ v16bf frag_b_glb(const unsigned int* __restrict__ Bpk,
                                            int kpairBase, int col16, int lane) {
    BF16Frag f;
    const unsigned int* p = Bpk + (size_t)(kpairBase + ((lane >> 4) << 2)) * NN + col16 + (lane & 15);
#pragma unroll
    for (int v = 0; v < 8; ++v)
        f.u[v] = p[(size_t)(((v & 4) ? 8 : 0) + (v & 3)) * NN];
    return f.v;
}

// ---- kernel 1: pack W0 into pair-packed bf16 dwords ------------------------
__global__ void wprep(const float* __restrict__ weights, unsigned int* __restrict__ W0pk) {
    int idx = blockIdx.x * blockDim.x + threadIdx.x;   // 2048 = 32 kpairs x 64 cols
    if (idx < 2048) {
        int kp = idx >> 6, c = idx & 63;
        unsigned int lo = f2bf(weights[(2 * kp)     * 64 + c]);
        unsigned int hi = f2bf(weights[(2 * kp + 1) * 64 + c]);
        W0pk[idx] = lo | (hi << 16);
    }
}

// ---- kernel 2: S = softmax(relu(adj @ emb)); store row-major + pair-packed -
__global__ __launch_bounds__(256) void supports_softmax(const float* __restrict__ adj,
                                                        const float* __restrict__ emb,
                                                        unsigned short* __restrict__ S,
                                                        unsigned int* __restrict__ Spk) {
    __shared__ float arow[16];
    __shared__ float red[256];
    int row = blockIdx.x, tid = threadIdx.x;
    if (tid < 16) arow[tid] = adj[row * 16 + tid];
    __syncthreads();

    float z[16];
    float lmax = 0.f;                                   // relu floor
#pragma unroll
    for (int i = 0; i < 16; ++i) {
        int j = tid + i * 256;
        float s = 0.f;
#pragma unroll
        for (int e = 0; e < 16; ++e) s = fmaf(arow[e], emb[e * NN + j], s);
        s = fmaxf(s, 0.f);
        z[i] = s;
        lmax = fmaxf(lmax, s);
    }
    red[tid] = lmax; __syncthreads();
    for (int off = 128; off > 0; off >>= 1) {
        if (tid < off) red[tid] = fmaxf(red[tid], red[tid + off]);
        __syncthreads();
    }
    float m = red[0];
    __syncthreads();

    float lsum = 0.f;
#pragma unroll
    for (int i = 0; i < 16; ++i) { z[i] = __expf(z[i] - m); lsum += z[i]; }
    red[tid] = lsum; __syncthreads();
    for (int off = 128; off > 0; off >>= 1) {
        if (tid < off) red[tid] += red[tid + off];
        __syncthreads();
    }
    float inv = 1.0f / red[0];
    unsigned short* Spk16 = (unsigned short*)Spk;
#pragma unroll
    for (int i = 0; i < 16; ++i) {
        int j = tid + i * 256;
        unsigned short bv = f2bf(z[i] * inv);
        S[(size_t)row * NN + j] = bv;                               // A-side, row-major
        Spk16[((size_t)(row >> 1) * NN + j) * 2 + (row & 1)] = bv;  // B-side, pair-packed
    }
}

// ---- kernel 3: Xb row-major bf16; Z1pk/Z2pk = pair-packed bf16 of X@W1/2 ---
__global__ __launch_bounds__(256) void zprep(const float* __restrict__ x,
                                             const float* __restrict__ weights,
                                             unsigned short* __restrict__ Xb,
                                             unsigned int* __restrict__ Z1pk,
                                             unsigned int* __restrict__ Z2pk) {
    __shared__ float xs[4][64];
    int tid  = threadIdx.x;
    int sub  = tid >> 6, c = tid & 63;
    int flat = blockIdx.x * 4 + sub;                    // flat = b*4096 + m
    int b = flat >> 12, m = flat & 4095;
    float xv = x[(size_t)flat * 64 + c];
    xs[sub][c] = xv;
    __syncthreads();

    const float* W1 = weights + 64 * 64;
    const float* W2 = weights + 2 * 64 * 64;
    float z1 = 0.f, z2 = 0.f;
#pragma unroll 8
    for (int i = 0; i < 64; ++i) {
        float xi = xs[sub][i];
        z1 = fmaf(xi, W1[i * 64 + c], z1);
        z2 = fmaf(xi, W2[i * 64 + c], z2);
    }
    int col = b * 64 + c;
    Xb[(size_t)m * NN + col] = f2bf(xv);
    size_t pk = ((size_t)(m >> 1) * NN + col) * 2 + (m & 1);
    ((unsigned short*)Z1pk)[pk] = f2bf(z1);
    ((unsigned short*)Z2pk)[pk] = f2bf(z2);
}

// ---- kernel 4: T2 = bf16( 2*(S@S) - I ) ------------------------------------
__global__ __launch_bounds__(256) void cheb2_gemm(const unsigned short* __restrict__ S,
                                                  const unsigned int* __restrict__ Spk,
                                                  unsigned short* __restrict__ T2) {
    __shared__ unsigned short As[2][128 * PA];
    int tid  = threadIdx.x;
    int lane = tid & 31, wid = tid >> 5;
    int wm = wid >> 2, wn = wid & 3;                    // 2x4 wave grid, 64x32 per wave
    int rowBlock = (blockIdx.x & 31) * 128;
    int colBlock = (blockIdx.x >> 5) * 128;

    v8f zero = {0.f, 0.f, 0.f, 0.f, 0.f, 0.f, 0.f, 0.f};
    v8f acc[4][2];
#pragma unroll
    for (int i = 0; i < 4; ++i)
#pragma unroll
        for (int j = 0; j < 2; ++j) acc[i][j] = zero;

#if USE_TDM
    if (wid == 0) tdm_load_tile_a(lds_off(As[0]), S + (size_t)rowBlock * NN);
#endif
    for (int s = 0; s < 128; ++s) {
        int kk = s * 32;
#if USE_TDM
        if (wid == 0) {
            if (s + 1 < 128) {
                tdm_load_tile_a(lds_off(As[(s + 1) & 1]), S + (size_t)rowBlock * NN + kk + 32);
                __builtin_amdgcn_s_wait_tensorcnt(1);   // current tile done (in-order)
            } else {
                __builtin_amdgcn_s_wait_tensorcnt(0);   // drain before last compute
            }
        }
        const unsigned short* Abuf = As[s & 1];
#else
        copy_tile_a(As[0], S, rowBlock, kk, tid);
        const unsigned short* Abuf = As[0];
#endif
        __syncthreads();
        v16bf a0 = frag_a_lds(Abuf, wm * 64,      lane);
        v16bf a1 = frag_a_lds(Abuf, wm * 64 + 16, lane);
        v16bf a2 = frag_a_lds(Abuf, wm * 64 + 32, lane);
        v16bf a3 = frag_a_lds(Abuf, wm * 64 + 48, lane);
#pragma unroll
        for (int j = 0; j < 2; ++j) {
            v16bf b = frag_b_glb(Spk, kk >> 1, colBlock + wn * 32 + j * 16, lane);
            acc[0][j] = wmma_bf16(a0, b, acc[0][j]);
            acc[1][j] = wmma_bf16(a1, b, acc[1][j]);
            acc[2][j] = wmma_bf16(a2, b, acc[2][j]);
            acc[3][j] = wmma_bf16(a3, b, acc[3][j]);
        }
        __syncthreads();
    }
#pragma unroll
    for (int i = 0; i < 4; ++i)
#pragma unroll
        for (int j = 0; j < 2; ++j)
#pragma unroll
            for (int v = 0; v < 8; ++v) {
                int gr = rowBlock + wm * 64 + i * 16 + v + ((lane >> 4) << 3);
                int gc = colBlock + wn * 32 + j * 16 + (lane & 15);
                float val = 2.0f * acc[i][j][v] - ((gr == gc) ? 1.0f : 0.0f);
                T2[(size_t)gr * NN + gc] = f2bf(val);
            }
}

// ---- kernel 5: out = X@W0 + S@Z1 + T2@Z2 + bias (f32 out [B,N,C]) ----------
__global__ __launch_bounds__(256) void aggregate_gemm(const unsigned short* __restrict__ S,
                                                      const unsigned short* __restrict__ T2,
                                                      const unsigned int* __restrict__ Z1pk,
                                                      const unsigned int* __restrict__ Z2pk,
                                                      const unsigned short* __restrict__ Xb,
                                                      const unsigned int*   __restrict__ W0pk,
                                                      const float* __restrict__ bias,
                                                      float* __restrict__ out) {
    __shared__ unsigned short As[2][128 * PA];
    int tid  = threadIdx.x;
    int lane = tid & 31, wid = tid >> 5;
    int wm = wid >> 2, wn = wid & 3;
    int rowBlock = (blockIdx.x & 31) * 128;
    int colBlock = (blockIdx.x >> 5) * 128;

    v8f zero = {0.f, 0.f, 0.f, 0.f, 0.f, 0.f, 0.f, 0.f};
    v8f acc[4][2];
#pragma unroll
    for (int i = 0; i < 4; ++i)
#pragma unroll
        for (int j = 0; j < 2; ++j) acc[i][j] = zero;

    // phases 0/1: S@Z1 then T2@Z2, K = 4096 each
    for (int ph = 0; ph < 2; ++ph) {
        const unsigned short* Ag  = ph ? T2   : S;
        const unsigned int*   Bpk = ph ? Z2pk : Z1pk;
#if USE_TDM
        if (wid == 0) tdm_load_tile_a(lds_off(As[0]), Ag + (size_t)rowBlock * NN);
#endif
        for (int s = 0; s < 128; ++s) {
            int kk = s * 32;
#if USE_TDM
            if (wid == 0) {
                if (s + 1 < 128) {
                    tdm_load_tile_a(lds_off(As[(s + 1) & 1]), Ag + (size_t)rowBlock * NN + kk + 32);
                    __builtin_amdgcn_s_wait_tensorcnt(1);
                } else {
                    __builtin_amdgcn_s_wait_tensorcnt(0);
                }
            }
            const unsigned short* Abuf = As[s & 1];
#else
            copy_tile_a(As[0], Ag, rowBlock, kk, tid);
            const unsigned short* Abuf = As[0];
#endif
            __syncthreads();
            v16bf a0 = frag_a_lds(Abuf, wm * 64,      lane);
            v16bf a1 = frag_a_lds(Abuf, wm * 64 + 16, lane);
            v16bf a2 = frag_a_lds(Abuf, wm * 64 + 32, lane);
            v16bf a3 = frag_a_lds(Abuf, wm * 64 + 48, lane);
#pragma unroll
            for (int j = 0; j < 2; ++j) {
                v16bf b = frag_b_glb(Bpk, kk >> 1, colBlock + wn * 32 + j * 16, lane);
                acc[0][j] = wmma_bf16(a0, b, acc[0][j]);
                acc[1][j] = wmma_bf16(a1, b, acc[1][j]);
                acc[2][j] = wmma_bf16(a2, b, acc[2][j]);
                acc[3][j] = wmma_bf16(a3, b, acc[3][j]);
            }
            __syncthreads();
        }
    }

    // phase 2: identity-k term, K = 64. Each wave's 32-col span lies in one batch b.
    int bIdx  = (colBlock + wn * 32) >> 6;
    int cBase = (wn & 1) * 32;
    int hf = lane >> 4, r = lane & 15;
#pragma unroll
    for (int s = 0; s < 2; ++s) {
        BF16Frag fa[4];
#pragma unroll
        for (int i = 0; i < 4; ++i)
#pragma unroll
            for (int v = 0; v < 8; ++v) {
                int kb  = ((v & 4) ? 16 : 0) + hf * 8 + 2 * (v & 3);
                int col = bIdx * 64 + s * 32 + kb;
                fa[i].u[v] = *(const unsigned int*)(Xb + (size_t)(rowBlock + wm * 64 + i * 16 + r) * NN + col);
            }
#pragma unroll
        for (int j = 0; j < 2; ++j) {
            BF16Frag fb;
            int c = cBase + j * 16 + (lane & 15);
#pragma unroll
            for (int v = 0; v < 8; ++v)
                fb.u[v] = W0pk[(s * 16 + ((v & 4) ? 8 : 0) + hf * 4 + (v & 3)) * 64 + c];
#pragma unroll
            for (int i = 0; i < 4; ++i)
                acc[i][j] = wmma_bf16(fa[i].v, fb.v, acc[i][j]);
        }
    }

    // epilogue: bias add, scatter to [B, N, C] f32
#pragma unroll
    for (int i = 0; i < 4; ++i)
#pragma unroll
        for (int j = 0; j < 2; ++j)
#pragma unroll
            for (int v = 0; v < 8; ++v) {
                int gr = rowBlock + wm * 64 + i * 16 + v + ((lane >> 4) << 3);
                int gc = colBlock + wn * 32 + j * 16 + (lane & 15);
                int b = gc >> 6, c = gc & 63;
                out[((size_t)b * NN + gr) * 64 + c] = acc[i][j][v] + bias[c];
            }
}

extern "C" void kernel_launch(void* const* d_in, const int* in_sizes, int n_in,
                              void* d_out, int out_size, void* d_ws, size_t ws_size,
                              hipStream_t stream) {
    (void)in_sizes; (void)n_in; (void)out_size; (void)ws_size;
    const float* x       = (const float*)d_in[0];   // [64, 4096, 64]
    const float* adj     = (const float*)d_in[1];   // [4096, 16]
    const float* emb     = (const float*)d_in[2];   // [16, 4096]
    const float* weights = (const float*)d_in[3];   // [3, 64, 64]
    const float* bias    = (const float*)d_in[4];   // [64]
    float* out = (float*)d_out;                     // [64, 4096, 64]

    const size_t MAT = (size_t)NN * NN * sizeof(unsigned short);  // 32 MB
    char* ws = (char*)d_ws;
    unsigned short* S    = (unsigned short*)(ws);
    unsigned int*   Spk  = (unsigned int*)  (ws + 1 * MAT);
    unsigned short* T2   = (unsigned short*)(ws + 2 * MAT);
    unsigned short* Xb   = (unsigned short*)(ws + 3 * MAT);
    unsigned int*   Z1pk = (unsigned int*)  (ws + 4 * MAT);
    unsigned int*   Z2pk = (unsigned int*)  (ws + 5 * MAT);
    unsigned int*   W0pk = (unsigned int*)  (ws + 6 * MAT);       // 8 KB

    wprep<<<8, 256, 0, stream>>>(weights, W0pk);
    supports_softmax<<<NN, 256, 0, stream>>>(adj, emb, S, Spk);
    zprep<<<(64 * NN) / 4, 256, 0, stream>>>(x, weights, Xb, Z1pk, Z2pk);
    cheb2_gemm<<<1024, 256, 0, stream>>>(S, Spk, T2);
    aggregate_gemm<<<1024, 256, 0, stream>>>(S, T2, Z1pk, Z2pk, Xb, W0pk, bias, out);
}
